// ChannelAttention3D_66606352826448
// MI455X (gfx1250) — matrix operand order repaired and checked by reading, not compiled, not run on hardware
//
#include <hip/hip_runtime.h>
#include <hip/hip_bf16.h>
#include <math.h>

// ---------------------------------------------------------------------------
// ChannelAttention3D on MI455X (gfx1250, wave32, WMMA 16x16x32 f16->f32)
//
// Memory-bound: 537 MB traffic vs 8.6 GFLOP -> target is one streaming pass
// over q,k (Gram), one over v (+out write). WMMA makes the 32768-deep channel
// contractions free. This revision vectorizes all global traffic to b128
// (4 contiguous channels per thread) and packs LDS staging stores to b64,
// cutting VMEM/DS issue 4x so instruction issue can't cap the 23.3 TB/s HBM.
// ---------------------------------------------------------------------------

typedef __attribute__((ext_vector_type(16))) _Float16 v16h;
typedef __attribute__((ext_vector_type(8)))  float    v8f;

union V16H { v16h v; unsigned int u[8]; };

static constexpr int BATCH = 4;
static constexpr int DHW   = 32768;
static constexpr int C     = 256;
static constexpr int H     = 4;
static constexpr int CH    = 64;           // channels per head

// ----------------------------- init --------------------------------------
__global__ void zero_f32(float* __restrict__ p, int n) {
    int i = blockIdx.x * 256 + threadIdx.x;
    if (i < n) p[i] = 0.0f;
}

// ----------------------- phase 1: S = Q^T K ------------------------------
// grid: 16 (b,h) * 32 chunks = 512 blocks, 256 threads (8 waves)
// each block: 1024 tokens, K-steps of 32; LDS tiles stored transposed [c][n]
static constexpr int P1_CHUNK = 1024;
static constexpr int P1_STEPS = P1_CHUNK / 32;

__global__ __launch_bounds__(256) void gram_qk(
    const float* __restrict__ qg, const float* __restrict__ kg,
    const float* __restrict__ qw, const float* __restrict__ qb,
    const float* __restrict__ kw, const float* __restrict__ kb,
    float* __restrict__ S)
{
    __shared__ _Float16 lq[CH][34];   // [channel][token] (transposed), even stride
    __shared__ _Float16 lk[CH][34];

    const int bh    = blockIdx.x >> 5;
    const int chunk = blockIdx.x & 31;
    const int b = bh >> 2, h = bh & 3;
    const int t = threadIdx.x;
    const int wave = t >> 5, lane = t & 31;
    const int m  = lane & 15;          // M / N index within tile
    const int hi = lane >> 4;          // lane half selects K group

    // wave -> one c-tile, two d-tiles (A reused across both WMMAs)
    const int ct  = wave >> 1;
    const int dt0 = (wave & 1) * 2;

    v8f acc0 = {}; v8f acc1 = {};

    const int n0 = chunk * P1_CHUNK;
    const size_t base = (size_t)b * DHW * C + (size_t)h * CH;

    // staging: each thread owns 4 contiguous channels -> b128 global loads
    const int c4 = (t & 15) * 4;       // 0,4,...,60
    const int nb = t >> 4;             // 0..15
    const float4 qwv = *(const float4*)&qw[h * CH + c4];
    const float4 qbv = *(const float4*)&qb[h * CH + c4];
    const float4 kwv = *(const float4*)&kw[h * CH + c4];
    const float4 kbv = *(const float4*)&kb[h * CH + c4];

    for (int s = 0; s < P1_STEPS; ++s) {
        const int ns = n0 + s * 32;
        __syncthreads();
#pragma unroll
        for (int i = 0; i < 2; ++i) {
            const int nl = nb + i * 16;                     // 0..31
            const size_t g = base + (size_t)(ns + nl) * C + c4;
            const float4 qv = *(const float4*)&qg[g];
            const float4 kv = *(const float4*)&kg[g];
            lq[c4 + 0][nl] = (_Float16)(qv.x * qwv.x + qbv.x);
            lq[c4 + 1][nl] = (_Float16)(qv.y * qwv.y + qbv.y);
            lq[c4 + 2][nl] = (_Float16)(qv.z * qwv.z + qbv.z);
            lq[c4 + 3][nl] = (_Float16)(qv.w * qwv.w + qbv.w);
            lk[c4 + 0][nl] = (_Float16)(kv.x * kwv.x + kbv.x);
            lk[c4 + 1][nl] = (_Float16)(kv.y * kwv.y + kbv.y);
            lk[c4 + 2][nl] = (_Float16)(kv.z * kwv.z + kbv.z);
            lk[c4 + 3][nl] = (_Float16)(kv.w * kwv.w + kbv.w);
        }
        __syncthreads();

        V16H A, B0, B1;
#pragma unroll
        for (int r = 0; r < 8; ++r) {
            const int kA = (r < 4 ? 2 * r : 2 * r + 8) + hi * 8;   // A-matrix K map
            const int kB = (r < 4 ? 2 * r : 2 * r + 8) + hi * 8;   // B-matrix K map
            A.u[r]  = *(const unsigned int*)&lq[ct * 16 + m][kA];
            B0.u[r] = *(const unsigned int*)&lk[dt0 * 16 + m][kB];
            B1.u[r] = *(const unsigned int*)&lk[(dt0 + 1) * 16 + m][kB];
        }
        acc0 = __builtin_amdgcn_wmma_f32_16x16x32_f16(false, A.v, false, B0.v,
                                                      (short)0, acc0, false, false);
        acc1 = __builtin_amdgcn_wmma_f32_16x16x32_f16(false, A.v, false, B1.v,
                                                      (short)0, acc1, false, false);
    }

    // D layout: VGPR r, lane: M = r + hi*8, N = lane%16
    float* Sb = S + (size_t)bh * CH * CH;
#pragma unroll
    for (int r = 0; r < 8; ++r) {
        const int c = ct * 16 + r + hi * 8;
        atomicAdd(&Sb[c * CH + dt0 * 16 + m],       acc0[r]);
        atomicAdd(&Sb[c * CH + (dt0 + 1) * 16 + m], acc1[r]);
    }
}

// ----------------------- phase 2: row softmax ----------------------------
// 16 blocks (b,h), 64 threads: one 64-wide row each. scale = C^-0.5 = 1/16
__global__ void softmax64(const float* __restrict__ S, float* __restrict__ A)
{
    const int bh = blockIdx.x;
    const int c  = threadIdx.x;
    const float4* row  = (const float4*)(S + ((size_t)bh * 64 + c) * 64);
    float4*       arow = (float4*)(A + ((size_t)bh * 64 + c) * 64);
    float4 vals[16];
    float mx = -3.0e38f;
#pragma unroll
    for (int d = 0; d < 16; ++d) {
        float4 x = row[d];
        x.x *= 0.0625f; x.y *= 0.0625f; x.z *= 0.0625f; x.w *= 0.0625f;
        vals[d] = x;
        mx = fmaxf(mx, fmaxf(fmaxf(x.x, x.y), fmaxf(x.z, x.w)));
    }
    float sum = 0.0f;
#pragma unroll
    for (int d = 0; d < 16; ++d) {
        vals[d].x = __expf(vals[d].x - mx); sum += vals[d].x;
        vals[d].y = __expf(vals[d].y - mx); sum += vals[d].y;
        vals[d].z = __expf(vals[d].z - mx); sum += vals[d].z;
        vals[d].w = __expf(vals[d].w - mx); sum += vals[d].w;
    }
    const float inv = 1.0f / sum;
#pragma unroll
    for (int d = 0; d < 16; ++d) {
        float4 x = vals[d];
        x.x *= inv; x.y *= inv; x.z *= inv; x.w *= inv;
        arow[d] = x;
    }
}

// ----------------------- phase 3: Out = V * Att^T ------------------------
// grid: 16 (b,h) * 256 chunks = 4096 blocks, 256 threads (8 waves)
// each block: 128 tokens; wave w owns token-tile w; loops 4 c-tiles
static constexpr int P3_CHUNK = 128;
static constexpr int LVS = 72;          // row stride (halves): 144 B, 8B-aligned

__global__ __launch_bounds__(256) void av_out(
    const float* __restrict__ vg,
    const float* __restrict__ vw, const float* __restrict__ vb,
    const float* __restrict__ att,
    const float* __restrict__ pw, const float* __restrict__ pb,
    float* __restrict__ out)
{
    __shared__ _Float16 lv[P3_CHUNK][LVS];   // [token][d]  (natural row-major)
    __shared__ _Float16 la[64][LVS];         // [c][d]      (att rows = B columns)

    const int bh    = blockIdx.x >> 8;
    const int chunk = blockIdx.x & 255;
    const int b = bh >> 2, h = bh & 3;
    const int t = threadIdx.x;
    const int wave = t >> 5, lane = t & 31;
    const int m  = lane & 15;
    const int hi = lane >> 4;

    const int n0 = chunk * P3_CHUNK;
    const size_t base = (size_t)b * DHW * C + (size_t)h * CH;

    const int c4 = (t & 15) * 4;       // 4 contiguous d-channels per thread
    const int nb = t >> 4;             // 0..15

    union PK4 { _Float16 hh[4]; uint2 u2; };

    // stage att (4096 f32 -> f16), b128 loads + b64 LDS stores
    const float* ab = att + (size_t)bh * 64 * 64;
#pragma unroll
    for (int i = 0; i < 4; ++i) {
        const int cl = nb + i * 16;
        const float4 x = *(const float4*)&ab[cl * 64 + c4];
        PK4 p;
        p.hh[0] = (_Float16)x.x; p.hh[1] = (_Float16)x.y;
        p.hh[2] = (_Float16)x.z; p.hh[3] = (_Float16)x.w;
        *(uint2*)&la[cl][c4] = p.u2;
    }
    // stage V tile (128x64 f32 -> f16 with affine)
    {
        const float4 vwv = *(const float4*)&vw[h * CH + c4];
        const float4 vbv = *(const float4*)&vb[h * CH + c4];
#pragma unroll
        for (int i = 0; i < 8; ++i) {
            const int nl = nb + i * 16;
            const float4 x = *(const float4*)&vg[base + (size_t)(n0 + nl) * C + c4];
            PK4 p;
            p.hh[0] = (_Float16)(x.x * vwv.x + vbv.x);
            p.hh[1] = (_Float16)(x.y * vwv.y + vbv.y);
            p.hh[2] = (_Float16)(x.z * vwv.z + vbv.z);
            p.hh[3] = (_Float16)(x.w * vwv.w + vbv.w);
            *(uint2*)&lv[nl][c4] = p.u2;
        }
    }
    __syncthreads();

    // A tiles (V rows) for this wave's token-tile, both K-halves (d 0-31, 32-63)
    V16H A0, A1;
#pragma unroll
    for (int r = 0; r < 8; ++r) {
        const int kA = (r < 4 ? 2 * r : 2 * r + 8) + hi * 8;
        A0.u[r] = *(const unsigned int*)&lv[wave * 16 + m][kA];
        A1.u[r] = *(const unsigned int*)&lv[wave * 16 + m][kA + 32];
    }

#pragma unroll
    for (int ctile = 0; ctile < 4; ++ctile) {
        V16H B0, B1;
#pragma unroll
        for (int r = 0; r < 8; ++r) {
            const int kB = (r < 4 ? 2 * r : 2 * r + 8) + hi * 8;
            B0.u[r] = *(const unsigned int*)&la[ctile * 16 + m][kB];
            B1.u[r] = *(const unsigned int*)&la[ctile * 16 + m][kB + 32];
        }
        v8f acc = {};
        acc = __builtin_amdgcn_wmma_f32_16x16x32_f16(false, A0.v, false, B0.v,
                                                     (short)0, acc, false, false);
        acc = __builtin_amdgcn_wmma_f32_16x16x32_f16(false, A1.v, false, B1.v,
                                                     (short)0, acc, false, false);
        // store with fused output affine; D: M = token = r + hi*8, N = c = lane%16
        const int cl = ctile * 16 + m;
        const int chn = h * CH + cl;
        const float pwv = pw[chn], pbv = pb[chn];
#pragma unroll
        for (int r = 0; r < 8; ++r) {
            const int nl = wave * 16 + r + hi * 8;
            out[base + (size_t)(n0 + nl) * C + cl] = acc[r] * pwv + pbv;
        }
    }
}

// ---------------------------------------------------------------------------
extern "C" void kernel_launch(void* const* d_in, const int* in_sizes, int n_in,
                              void* d_out, int out_size, void* d_ws, size_t ws_size,
                              hipStream_t stream) {
    const float* q  = (const float*)d_in[0];
    const float* k  = (const float*)d_in[1];
    const float* v  = (const float*)d_in[2];
    const float* qw = (const float*)d_in[3];
    const float* qb = (const float*)d_in[4];
    const float* kw = (const float*)d_in[5];
    const float* kb = (const float*)d_in[6];
    const float* vw = (const float*)d_in[7];
    const float* vb = (const float*)d_in[8];
    const float* pw = (const float*)d_in[9];
    const float* pb = (const float*)d_in[10];

    float* S = (float*)d_ws;                     // 16*64*64 f32 = 256 KB
    float* A = S + (size_t)BATCH * H * CH * CH;  // 16*64*64 f32 = 256 KB

    const int sElems = BATCH * H * CH * CH;      // 65536
    zero_f32<<<(sElems + 255) / 256, 256, 0, stream>>>(S, sElems);
    gram_qk<<<BATCH * H * (DHW / P1_CHUNK), 256, 0, stream>>>(q, k, qw, qb, kw, kb, S);
    softmax64<<<BATCH * H, 64, 0, stream>>>(S, A);
    av_out<<<BATCH * H * (DHW / P3_CHUNK), 256, 0, stream>>>(v, vw, vb, A, pw, pb,
                                                             (float*)d_out);
}
